// HybridModel_63178968924454
// MI455X (gfx1250) — matrix operand order repaired and checked by reading, not compiled, not run on hardware
//
#include <hip/hip_runtime.h>
#include <hip/hip_bf16.h>
#include <math.h>

typedef __attribute__((ext_vector_type(16))) _Float16 v16h;
typedef __attribute__((ext_vector_type(8)))  _Float16 v8h;
typedef __attribute__((ext_vector_type(8)))  float    v8f;

namespace {
constexpr int kV = 32000, kD = 128, kB = 4, kS = 1024, kNH = 4, kHD = 32, kFF = 256;
constexpr int kBS  = kB * kS;              // 4096 token rows
constexpr long kBSD = (long)kBS * kD;      // 524288
constexpr int k3D = 3 * kD;                // 384
}

// ---------------- elementwise / gather / norm kernels ----------------

__global__ void embed_kernel(const int* __restrict__ x, const float* __restrict__ emb,
                             float* __restrict__ h) {
  long i = (long)blockIdx.x * blockDim.x + threadIdx.x;
  if (i >= kBSD) return;
  long bs = i / kD; int d = (int)(i % kD);
  h[i] = emb[(long)x[bs] * kD + d];
}

__global__ void f32_to_f16_kernel(const float* __restrict__ x, _Float16* __restrict__ y, long n) {
  long i = (long)blockIdx.x * blockDim.x + threadIdx.x;
  if (i < n) y[i] = (_Float16)x[i];
}

// vT[b,h,hd,s] = qkv16[b,s, 2D + h*HD + hd]  (so attn@v B-fragments are contiguous)
__global__ void v_transpose_kernel(const _Float16* __restrict__ qkv16,
                                   _Float16* __restrict__ vT) {
  long i = (long)blockIdx.x * blockDim.x + threadIdx.x;
  if (i >= kBSD) return;
  int d = (int)(i % kD);
  long bs = i / kD;
  int b = (int)(bs / kS), s = (int)(bs % kS);
  int h = d / kHD, hd = d % kHD;
  vT[(((long)b * kNH + h) * kHD + hd) * kS + s] = qkv16[bs * k3D + 2 * kD + d];
}

__global__ void ln_kernel(const float* __restrict__ x, const float* __restrict__ g,
                          const float* __restrict__ b, float* __restrict__ y) {
  __shared__ float red[kD];
  int row = blockIdx.x, t = threadIdx.x;  // blockDim.x == kD == 128
  float v = x[(long)row * kD + t];
  red[t] = v; __syncthreads();
  for (int s = kD / 2; s > 0; s >>= 1) { if (t < s) red[t] += red[t + s]; __syncthreads(); }
  float mean = red[0] * (1.f / kD);
  __syncthreads();
  float dv = v - mean;
  red[t] = dv * dv; __syncthreads();
  for (int s = kD / 2; s > 0; s >>= 1) { if (t < s) red[t] += red[t + s]; __syncthreads(); }
  float inv = rsqrtf(red[0] * (1.f / kD) + 1e-5f);
  y[(long)row * kD + t] = dv * inv * g[t] + b[t];
}

__global__ void softmax_rows_kernel(const float* __restrict__ sc, _Float16* __restrict__ attn) {
  __shared__ float red[256];
  int row = blockIdx.x, t = threadIdx.x;  // blockDim.x == 256, row over kS
  const float* r = sc + (long)row * kS;
  float mx = -3.4e38f;
  for (int c = t; c < kS; c += 256) mx = fmaxf(mx, r[c]);
  red[t] = mx; __syncthreads();
  for (int s = 128; s > 0; s >>= 1) { if (t < s) red[t] = fmaxf(red[t], red[t + s]); __syncthreads(); }
  mx = red[0]; __syncthreads();
  float sum = 0.f;
  for (int c = t; c < kS; c += 256) sum += expf(r[c] - mx);
  red[t] = sum; __syncthreads();
  for (int s = 128; s > 0; s >>= 1) { if (t < s) red[t] += red[t + s]; __syncthreads(); }
  float inv = 1.f / red[0];
  _Float16* o = attn + (long)row * kS;
  for (int c = t; c < kS; c += 256) o[c] = (_Float16)(expf(r[c] - mx) * inv);
}

// Linear recurrence: 512 independent (b,d) chains, sequential over S.
__global__ void mamba_scan_kernel(const float* __restrict__ z, const float* __restrict__ bv,
                                  const float* __restrict__ gp, const float* __restrict__ lam,
                                  float* __restrict__ u) {
  int idx = blockIdx.x * blockDim.x + threadIdx.x;
  if (idx >= kB * kD) return;
  int b = idx / kD, d = idx % kD;
  float la = lam[d];
  float P = 1.f, Sacc = 0.f;
  long base = (long)b * kS * kD + d;
  for (int s = 0; s < kS; ++s) {
    long o = base + (long)s * kD;
    float delta = 1.f / (1.f + expf(-z[o]));
    float A = expf(-delta * la);
    float g = 1.f / (1.f + expf(-gp[o]));
    float a = g * A + 1e-6f;          // gated A + eps
    float bb = (1.f - g) * bv[o];     // gated b
    P *= a;                           // cumprod
    Sacc += bb / (P + 1e-6f);         // cumsum of b/P
    u[o] = P * Sacc;
  }
}

// ---------------- WMMA GEMM ----------------
// C[M,N] = act(alpha * A[M,K] x B(K,N) + bias) + addC
// A: f16 row-major, leading dim lda. B row-contiguous in K: B(k,n) = Bp[n*sn + k]
// (weights [N,K], k^T slices of qkv, pre-transposed vT).
// One wave computes a 16 x (16*TN) output strip; K stepped by 32 per WMMA.

__device__ __forceinline__ v16h make_a_frag(const _Float16* __restrict__ ap) {
  // ap already points at row base + k0 + 16*half; contiguous: e -> K=16*half+e
  v8h lo = *(const v8h*)(ap);
  v8h hi = *(const v8h*)(ap + 8);
  return __builtin_shufflevector(lo, hi, 0, 1, 2, 3, 4, 5, 6, 7,
                                 8, 9, 10, 11, 12, 13, 14, 15);
}

template <int TN>
__global__ void wmma_gemm_kernel(const _Float16* __restrict__ A, long lda,
                                 const _Float16* __restrict__ Bp, long sn,
                                 const float* __restrict__ bias,
                                 const float* __restrict__ addC, long ldadd,
                                 float* __restrict__ C, long ldc,
                                 int M, int N, int K, float alpha, int act) {
  int lane = threadIdx.x & 31;
  int wave = threadIdx.x >> 5;
  int wpb  = blockDim.x >> 5;
  int n0 = (blockIdx.x * wpb + wave) * 16 * TN;   // strip base (N % (16*TN) == 0 guaranteed)
  int m0 = blockIdx.y * 16;
  if (n0 >= N || m0 >= M) return;

  int half = lane >> 4, l15 = lane & 15;
  const _Float16* ap = A + (long)(m0 + l15) * lda + 16 * half;   // lo chunk base
  const _Float16* bp = Bp + (long)(n0 + l15) * sn + 16 * half;   // lanes 0-15: K lo, 16-31: +16

  v8f acc[TN];
#pragma unroll
  for (int t = 0; t < TN; ++t) acc[t] = v8f{};

  for (int k0 = 0; k0 < K; k0 += 32) {
    v16h a = make_a_frag(ap);                     // shared across the TN WMMAs
    __builtin_prefetch(bp + 32, 0, 1);            // next K-chunk (global_prefetch_b8)
#pragma unroll
    for (int t = 0; t < TN; ++t) {
      v16h b = *(const v16h*)(bp + (long)t * 16 * sn);  // contiguous 16 halfs, 32B aligned
      acc[t] = __builtin_amdgcn_wmma_f32_16x16x32_f16(false, a, false, b, (short)0, acc[t],
                                                      false, false);
    }
    ap += 32;
    bp += 32;
  }

#pragma unroll
  for (int t = 0; t < TN; ++t) {
    int col = n0 + 16 * t + l15;
    float bv = bias ? bias[col] : 0.f;
#pragma unroll
    for (int r = 0; r < 8; ++r) {
      int row = m0 + r + 8 * half;               // ISA f32 C/D 16x16 layout
      float v = acc[t][r] * alpha + bv;
      if (act == 1) v = 0.5f * v * (1.f + erff(v * 0.70710678118654752f));  // exact GELU
      if (addC) v += addC[(long)row * ldadd + col];
      C[(long)row * ldc + col] = v;
    }
  }
}

// ---------------- host-side helpers ----------------

static void launch_gemm(hipStream_t st, const _Float16* A, long lda,
                        const _Float16* Bp, long sn,
                        const float* bias, const float* addC, long ldadd,
                        float* C, long ldc, int M, int N, int K, float alpha, int act) {
  int TN = (N % 64 == 0) ? 4 : 2;                  // all N here are %64==0 except N=32
  int ntiles = N / (16 * TN);                      // strips along N
  int wpb = ntiles < 4 ? ntiles : 4;
  dim3 grid((unsigned)((ntiles + wpb - 1) / wpb), (unsigned)(M / 16));
  dim3 blk((unsigned)(32 * wpb));
  if (TN == 4)
    wmma_gemm_kernel<4><<<grid, blk, 0, st>>>(A, lda, Bp, sn, bias, addC, ldadd,
                                              C, ldc, M, N, K, alpha, act);
  else
    wmma_gemm_kernel<2><<<grid, blk, 0, st>>>(A, lda, Bp, sn, bias, addC, ldadd,
                                              C, ldc, M, N, K, alpha, act);
}

static void cvt(hipStream_t st, const float* x, _Float16* y, long n) {
  f32_to_f16_kernel<<<dim3((unsigned)((n + 255) / 256)), 256, 0, st>>>(x, y, n);
}

extern "C" void kernel_launch(void* const* d_in, const int* in_sizes, int n_in,
                              void* d_out, int out_size, void* d_ws, size_t ws_size,
                              hipStream_t stream) {
  (void)in_sizes; (void)n_in; (void)out_size; (void)ws_size;
  const int*   x     = (const int*)d_in[0];
  const float* embed = (const float*)d_in[1];
  const float* in_w  = (const float*)d_in[2];  const float* in_b  = (const float*)d_in[3];
  const float* out_w = (const float*)d_in[4];  const float* out_b = (const float*)d_in[5];
  const float* ln1g  = (const float*)d_in[6];  const float* ln1b  = (const float*)d_in[7];
  const float* ln2g  = (const float*)d_in[8];  const float* ln2b  = (const float*)d_in[9];
  const float* ln3g  = (const float*)d_in[10]; const float* ln3b  = (const float*)d_in[11];
  const float* mBw   = (const float*)d_in[12]; const float* mBb   = (const float*)d_in[13];
  const float* mCw   = (const float*)d_in[14]; const float* mCb   = (const float*)d_in[15];
  const float* mgw   = (const float*)d_in[16]; const float* mgb   = (const float*)d_in[17];
  const float* lam   = (const float*)d_in[18];
  const float* ff1w  = (const float*)d_in[19]; const float* ff1b  = (const float*)d_in[20];
  const float* ff2w  = (const float*)d_in[21]; const float* ff2b  = (const float*)d_in[22];
  const float* headw = (const float*)d_in[23]; const float* headb = (const float*)d_in[24];
  float* out = (float*)d_out;

  // ---- workspace bump allocator (256B-aligned blocks) ----
  char* wp = (char*)d_ws;
  auto alloc = [&](size_t bytes) -> void* {
    void* r = (void*)wp; wp += (bytes + 255) & ~(size_t)255; return r;
  };
  _Float16* w_in16   = (_Float16*)alloc(sizeof(_Float16) * (size_t)k3D * kD);
  _Float16* w_out16  = (_Float16*)alloc(sizeof(_Float16) * (size_t)kD * kD);
  _Float16* w_mB16   = (_Float16*)alloc(sizeof(_Float16) * (size_t)kD * kD);
  _Float16* w_mC16   = (_Float16*)alloc(sizeof(_Float16) * (size_t)kD * kD);
  _Float16* w_mg16   = (_Float16*)alloc(sizeof(_Float16) * (size_t)kD * kD);
  _Float16* w_ff116  = (_Float16*)alloc(sizeof(_Float16) * (size_t)kFF * kD);
  _Float16* w_ff216  = (_Float16*)alloc(sizeof(_Float16) * (size_t)kD * kFF);
  _Float16* w_head16 = (_Float16*)alloc(sizeof(_Float16) * (size_t)kV * kD);
  float*    hA    = (float*)alloc(sizeof(float) * kBSD);
  float*    hB    = (float*)alloc(sizeof(float) * kBSD);
  float*    y     = (float*)alloc(sizeof(float) * kBSD);           // LN outputs (reused)
  _Float16* act16 = (_Float16*)alloc(sizeof(_Float16) * kBSD);     // f16 activation staging
  float*    qkv   = (float*)alloc(sizeof(float) * (size_t)kBS * k3D);
  _Float16* qkv16 = (_Float16*)alloc(sizeof(_Float16) * (size_t)kBS * k3D);
  _Float16* vT    = (_Float16*)alloc(sizeof(_Float16) * kBSD);     // v transposed [B,H,HD,S]
  float*    scores = (float*)alloc(sizeof(float) * (size_t)kS * kS);       // reused per (b,h)
  _Float16* attn16 = (_Float16*)alloc(sizeof(_Float16) * (size_t)kS * kS); // reused per (b,h)
  float*    ao    = (float*)alloc(sizeof(float) * kBSD);
  float*    bvec  = (float*)alloc(sizeof(float) * kBSD);
  float*    gpre  = (float*)alloc(sizeof(float) * kBSD);
  float*    uu    = (float*)alloc(sizeof(float) * kBSD);
  float*    ffa   = (float*)alloc(sizeof(float) * (size_t)kBS * kFF);
  _Float16* ffa16 = (_Float16*)alloc(sizeof(_Float16) * (size_t)kBS * kFF);

  // ---- convert all weights to f16 (deterministic, every call) ----
  cvt(stream, in_w,  w_in16,  (long)k3D * kD);
  cvt(stream, out_w, w_out16, (long)kD * kD);
  cvt(stream, mBw,   w_mB16,  (long)kD * kD);
  cvt(stream, mCw,   w_mC16,  (long)kD * kD);
  cvt(stream, mgw,   w_mg16,  (long)kD * kD);
  cvt(stream, ff1w,  w_ff116, (long)kFF * kD);
  cvt(stream, ff2w,  w_ff216, (long)kD * kFF);
  cvt(stream, headw, w_head16,(long)kV * kD);

  // ---- embedding gather ----
  embed_kernel<<<dim3((unsigned)((kBSD + 255) / 256)), 256, 0, stream>>>(x, embed, hA);

  // ---- attention block ----
  ln_kernel<<<kBS, kD, 0, stream>>>(hA, ln1g, ln1b, y);
  cvt(stream, y, act16, kBSD);
  launch_gemm(stream, act16, kD, w_in16, kD, in_b, nullptr, 0,
              qkv, k3D, kBS, k3D, kD, 1.f, 0);                    // qkv = LN(h) @ Win^T + b
  cvt(stream, qkv, qkv16, (long)kBS * k3D);
  v_transpose_kernel<<<dim3((unsigned)((kBSD + 255) / 256)), 256, 0, stream>>>(qkv16, vT);
  const float kInvSqrtHD = 0.17677669529663688f;                  // 1/sqrt(32)
  for (int b = 0; b < kB; ++b) {
    for (int h = 0; h < kNH; ++h) {
      const _Float16* qA = qkv16 + (long)b * kS * k3D + h * kHD;          // q[m,k]: lda=3D
      const _Float16* kB_ = qkv16 + (long)b * kS * k3D + kD + h * kHD;    // k^T rows: sn=3D
      const _Float16* vB = vT + ((long)b * kNH + h) * kHD * kS;           // vT rows: sn=S
      launch_gemm(stream, qA, k3D, kB_, k3D, nullptr, nullptr, 0,
                  scores, kS, kS, kS, kHD, kInvSqrtHD, 0);        // scores = q k^T / sqrt(hd)
      softmax_rows_kernel<<<kS, 256, 0, stream>>>(scores, attn16);
      launch_gemm(stream, attn16, kS, vB, kS, nullptr, nullptr, 0,
                  ao + (long)b * kS * kD + h * kHD, kD, kS, kHD, kS, 1.f, 0);  // ao = attn v
    }
  }
  cvt(stream, ao, act16, kBSD);
  launch_gemm(stream, act16, kD, w_out16, kD, out_b, hA, kD,
              hB, kD, kBS, kD, kD, 1.f, 0);                       // h = h + ao @ Wout^T + b

  // ---- Mamba scan block ----
  ln_kernel<<<kBS, kD, 0, stream>>>(hB, ln2g, ln2b, y);           // z
  cvt(stream, y, act16, kBSD);
  launch_gemm(stream, act16, kD, w_mB16, kD, mBb, nullptr, 0,
              bvec, kD, kBS, kD, kD, 1.f, 0);                     // b = z @ mB^T + b
  launch_gemm(stream, act16, kD, w_mg16, kD, mgb, nullptr, 0,
              gpre, kD, kBS, kD, kD, 1.f, 0);                     // g_pre = z @ mg^T + b
  mamba_scan_kernel<<<2, 256, 0, stream>>>(y, bvec, gpre, lam, uu);
  cvt(stream, uu, act16, kBSD);
  launch_gemm(stream, act16, kD, w_mC16, kD, mCb, hB, kD,
              hA, kD, kBS, kD, kD, 1.f, 0);                       // h = h + (P*S) @ mC^T + b

  // ---- FFN block ----
  ln_kernel<<<kBS, kD, 0, stream>>>(hA, ln3g, ln3b, y);
  cvt(stream, y, act16, kBSD);
  launch_gemm(stream, act16, kD, w_ff116, kD, ff1b, nullptr, 0,
              ffa, kFF, kBS, kFF, kD, 1.f, 1);                    // gelu(f @ W1^T + b)
  cvt(stream, ffa, ffa16, (long)kBS * kFF);
  launch_gemm(stream, ffa16, kFF, w_ff216, kFF, ff2b, hA, kD,
              hB, kD, kBS, kD, kFF, 1.f, 0);                      // h = h + ff @ W2^T + b

  // ---- LM head (store-bandwidth bound: 524 MB of logits) ----
  cvt(stream, hB, act16, kBSD);
  launch_gemm(stream, act16, kD, w_head16, kD, headb, nullptr, 0,
              out, kV, kBS, kV, kD, 1.f, 0);
}
